// VarAttention_5574867550628
// MI455X (gfx1250) — compile-verified, matrix operand order checked by reading, not run on hardware
//
#include <hip/hip_runtime.h>
#include <hip/hip_bf16.h>
#include <stdint.h>

// ---------------------------------------------------------------- types ----
typedef __bf16 bf16;
typedef __attribute__((ext_vector_type(8)))  __bf16 v8bf;
typedef __attribute__((ext_vector_type(16))) __bf16 v16bf;
typedef __attribute__((ext_vector_type(8)))  float  v8f;
typedef __attribute__((ext_vector_type(4)))  unsigned int v4u;
typedef __attribute__((ext_vector_type(4)))  int    v4i;
typedef __attribute__((ext_vector_type(8)))  int    v8i;

static __device__ inline v16bf cat8(v8bf lo, v8bf hi) {
    return __builtin_shufflevector(lo, hi, 0,1,2,3,4,5,6,7,8,9,10,11,12,13,14,15);
}
static __device__ inline v8f wmma_bf16(v16bf a, v16bf b, v8f c) {
    // D = A(16x32 bf16) * B(32x16 bf16) + C(16x16 f32)
    return __builtin_amdgcn_wmma_f32_16x16x32_bf16(false, a, false, b, (short)0, c, false, false);
}

// Problem constants
#define Bc   4
#define Nt   96      // tokens
#define Pp   64      // patches
#define Cc   1024
#define Hh   16
#define HD   64
#define Mrows (Bc*Nt*Pp)   // 24576

// ------------------------------------------------------------- TDM load ----
// 2D tile load Global -> LDS via Tensor Data Mover (D# per CDNA5 ISA).
// tile_x elements (bf16) per row, tile_y rows, row stride = stride_elems.
// pad: +4 DWORDs (8 halfs) of LDS padding after every row (row must be a
// multiple of 32 DWORDs for the chosen interval code).
template <int TILE_X, int TILE_Y, int PAD_IVL>   // PAD_IVL: 2^(c+1) DWORDs per row
static __device__ inline void tdm_load_2d(unsigned int lds_off, const bf16* g,
                                          unsigned int stride_elems)
{
    uint64_t ga = (uint64_t)(uintptr_t)g;
    v4u g0;
    g0[0] = 1u;                                         // count=1 (valid, user)
    g0[1] = lds_off;                                    // LDS byte address
    g0[2] = (unsigned int)ga;                           // global_addr[31:0]
    g0[3] = (unsigned int)((ga >> 32) & 0x1FFFFFFu) | (2u << 30);  // [56:32], type=2

    v8i g1;
    unsigned int w0 = (1u << 16)                        // data_size = 1 (2 bytes)
                    | (1u << 20)                        // pad_enable
                    | ((unsigned)PAD_IVL << 22)         // pad_interval code
                    | (3u << 25);                       // pad_amount = 4 DWORDs (16B)
    unsigned int td0 = 0x40000000u, td1 = 0x40000000u;  // huge dims: no OOB clip
    g1[0] = (int)w0;
    g1[1] = (int)((td0 & 0xFFFFu) << 16);               // [63:48] = tensor_dim0 lo
    g1[2] = (int)((td0 >> 16) | ((td1 & 0xFFFFu) << 16));
    g1[3] = (int)((td1 >> 16) | ((unsigned)TILE_X << 16));   // [127:112] tile_dim0
    g1[4] = (int)((unsigned)TILE_Y);                    // [143:128] tile_dim1, dim2=0
    g1[5] = (int)stride_elems;                          // tensor_dim0_stride[31:0]
    g1[6] = 0;                                          // stride hi / dim1_stride lo
    g1[7] = 0;

    v4i z4 = {};
#if __clang_major__ >= 23
    v8i z8 = {};
    __builtin_amdgcn_tensor_load_to_lds(g0, g1, z4, z4, z8, 0);
#else
    __builtin_amdgcn_tensor_load_to_lds(g0, g1, z4, z4, 0);
#endif
}

// ------------------------------------------------- pool-over-P + convert ---
// x (B,N,P,C) f32 -> x_bf16 (same layout), x_pooled_bf16 (B*N, C) = mean over P
__global__ __launch_bounds__(256) void pool_convert_kernel(
    const float* __restrict__ x, bf16* __restrict__ xb, bf16* __restrict__ xp)
{
    int bn = blockIdx.y;                          // 0..383
    int c  = blockIdx.x * 256 + threadIdx.x;      // 0..1023
    size_t base = (size_t)bn * Pp * Cc;
    float s = 0.f;
#pragma unroll 8
    for (int p = 0; p < Pp; ++p) {
        float v = x[base + (size_t)p * Cc + c];
        s += v;
        xb[base + (size_t)p * Cc + c] = (bf16)v;
    }
    xp[(size_t)bn * Cc + c] = (bf16)(s * (1.0f / Pp));
}

// ------------------------------------------- transpose + convert weights ---
// Wt[n][k] = (bf16) W[k][col0 + n]   (W row-major K x ldw), K cols out
__global__ __launch_bounds__(256) void transpose_conv_kernel(
    const float* __restrict__ W, int ldw, int col0, int K, bf16* __restrict__ Wt)
{
    __shared__ float t[32][33];
    int n0 = blockIdx.x * 32, k0 = blockIdx.y * 32;
    int tx = threadIdx.x, ty = threadIdx.y;       // (32, 8)
#pragma unroll
    for (int j = 0; j < 4; ++j)
        t[ty + j * 8][tx] = W[(size_t)(k0 + ty + j * 8) * ldw + col0 + n0 + tx];
    __syncthreads();
#pragma unroll
    for (int j = 0; j < 4; ++j)
        Wt[(size_t)(n0 + ty + j * 8) * K + k0 + tx] = (bf16)t[tx][ty + j * 8];
}

// ------------------------------------------------------- tiled WMMA GEMM ---
// C(MxN) = A(MxK) * Bt(NxK)^T ; bf16 in, f32 accumulate.
// Tile 128x128x64; TDM double-buffered tile loads (wave 0 issues DMA);
// 8 waves (2 along M x 4 along N), each wave owns 4x2 16x16 accumulators.
// MODE 0: f32 row-major (qk pooled proj)
// MODE 1: bf16 scatter into v_t[bh][e'=p*64+d][ntok]      (v projection)
// MODE 2: f32 row-major + bias                            (output proj)
#define LDSW 72   // halfs per LDS row: 64 data + 8 pad (TDM pad: 16B / 128B row)
template <int MODE>
__global__ __launch_bounds__(256) void gemm_tiled_kernel(
    const bf16* __restrict__ A, const bf16* __restrict__ Bt,
    int M, int N, int K,
    float* __restrict__ outF, bf16* __restrict__ outH,
    const float* __restrict__ bias)
{
    __shared__ bf16 sA[2][128 * LDSW];
    __shared__ bf16 sB[2][128 * LDSW];

    const int tid  = threadIdx.x;
    const int wave = tid >> 5, lane = tid & 31;
    const int l15  = lane & 15;
    const int kb   = (lane >> 4) * 8;             // K sub-offset per ISA layout
    const int wm   = wave & 1;                    // 0..1  (64 rows each)
    const int wn   = wave >> 1;                   // 0..3  (32 cols each)
    const int blockM = blockIdx.y * 128;
    const int blockN = blockIdx.x * 128;

    const unsigned int ldsA0 = (unsigned int)(uintptr_t)&sA[0][0];
    const unsigned int ldsA1 = (unsigned int)(uintptr_t)&sA[1][0];
    const unsigned int ldsB0 = (unsigned int)(uintptr_t)&sB[0][0];
    const unsigned int ldsB1 = (unsigned int)(uintptr_t)&sB[1][0];

    v8f acc[4][2];
#pragma unroll
    for (int i = 0; i < 4; ++i) { acc[i][0] = (v8f){}; acc[i][1] = (v8f){}; }

    // Prologue: DMA first K-tile into buffer 0 (one issue per workgroup).
    if (wave == 0) {
        tdm_load_2d<64, 128, 4>(ldsA0, A  + (size_t)blockM * K, (unsigned)K);
        tdm_load_2d<64, 128, 4>(ldsB0, Bt + (size_t)blockN * K, (unsigned)K);
    }

    int buf = 0;
    for (int kk = 0; kk < K; kk += 64) {
        if (wave == 0) {
            if (kk + 64 < K) {                     // prefetch next tile via TDM
                unsigned int la = buf ? ldsA0 : ldsA1;
                unsigned int lb = buf ? ldsB0 : ldsB1;
                tdm_load_2d<64, 128, 4>(la, A  + (size_t)blockM * K + kk + 64, (unsigned)K);
                tdm_load_2d<64, 128, 4>(lb, Bt + (size_t)blockN * K + kk + 64, (unsigned)K);
                __builtin_amdgcn_s_wait_tensorcnt(2);   // current buffer complete
            } else {
                __builtin_amdgcn_s_wait_tensorcnt(0);
            }
        }
        __syncthreads();

        const bf16* sAb = &sA[buf][0];
        const bf16* sBb = &sB[buf][0];
#pragma unroll
        for (int k2 = 0; k2 < 2; ++k2) {
            const int ko = k2 * 32;
            v16bf af[4], bfr[2];
#pragma unroll
            for (int ms = 0; ms < 4; ++ms) {
                int r = (wm * 64 + ms * 16 + l15) * LDSW + ko + kb;
                af[ms] = cat8(*(const v8bf*)(sAb + r),
                              *(const v8bf*)(sAb + r + 16));
            }
#pragma unroll
            for (int ns = 0; ns < 2; ++ns) {
                int c = (wn * 32 + ns * 16 + l15) * LDSW + ko + kb;
                bfr[ns] = cat8(*(const v8bf*)(sBb + c),
                               *(const v8bf*)(sBb + c + 16));
            }
#pragma unroll
            for (int ms = 0; ms < 4; ++ms)
#pragma unroll
                for (int ns = 0; ns < 2; ++ns)
                    acc[ms][ns] = wmma_bf16(af[ms], bfr[ns], acc[ms][ns]);
        }
        __syncthreads();
        buf ^= 1;
    }

    // ------------------------------------------------------------ epilogue
    const int hi8 = (lane >> 4) << 3;
#pragma unroll
    for (int ms = 0; ms < 4; ++ms) {
#pragma unroll
        for (int ns = 0; ns < 2; ++ns) {
            int mBase = blockM + wm * 64 + ms * 16;
            int nCol  = blockN + wn * 32 + ns * 16 + l15;
#pragma unroll
            for (int j = 0; j < 8; ++j) {
                int   m   = mBase + j + hi8;
                float val = acc[ms][ns][j];
                if constexpr (MODE == 0) {
                    outF[(size_t)m * N + nCol] = val;
                } else if constexpr (MODE == 1) {
                    // m = (b, ntok, p); nCol = (h, d) -> v_t[bh][p*64+d][ntok]
                    int b = m / (Nt * Pp), r = m - b * (Nt * Pp);
                    int ntok = r >> 6, p = r & 63;
                    int h = nCol >> 6,  d = nCol & 63;
                    outH[((size_t)(b * Hh + h) * (HD * Pp) + p * 64 + d) * Nt + ntok]
                        = (bf16)val;
                } else {
                    outF[(size_t)m * N + nCol] = val + bias[nCol];
                }
            }
        }
    }
}

// ------------------------------------------------ scores + softmax (f32) ---
// qk: (B*N, 2048) f32 : [0,1024)=q_pooled, [1024,2048)=k_pooled (h-major, d)
// attn: (B*H, 96, 96) bf16
__global__ __launch_bounds__(96) void scores_softmax_kernel(
    const float* __restrict__ qk, bf16* __restrict__ attn)
{
    __shared__ float ks[Nt][HD];
    __shared__ float es[Nt][Nt + 1];
    int bh = blockIdx.x, b = bh >> 4, h = bh & 15;
    int n  = threadIdx.x;
    const float* base = qk + (size_t)b * Nt * 2048;

    float qreg[HD];
#pragma unroll
    for (int d = 0; d < HD; ++d) {
        ks[n][d] = base[(size_t)n * 2048 + 1024 + h * HD + d];
        qreg[d]  = base[(size_t)n * 2048 + h * HD + d];
    }
    __syncthreads();

    float mx = -3.0e38f;
    for (int m = 0; m < Nt; ++m) {
        float s = 0.f;
#pragma unroll
        for (int d = 0; d < HD; ++d) s += qreg[d] * ks[m][d];
        s *= 0.125f;                  // 1/sqrt(hd), hd = 64
        es[n][m] = s;
        mx = fmaxf(mx, s);
    }
    float sum = 0.f;
    for (int m = 0; m < Nt; ++m) {
        float e = __expf(es[n][m] - mx);
        es[n][m] = e;
        sum += e;
    }
    float inv = 1.f / sum;
    for (int m = 0; m < Nt; ++m)
        attn[((size_t)bh * Nt + n) * Nt + m] = (bf16)(es[n][m] * inv);
}

// --------------------------------------------------- attn @ v (per b,h) ----
// A = attn[bh] (96x96) bf16 ; Bt = v_t[bh] (4096x96) bf16 (rows e'=p*64+d)
// All fragments hoisted first so loads pipeline ahead of the WMMA burst.
__global__ __launch_bounds__(256) void attn_v_kernel(
    const bf16* __restrict__ attn, const bf16* __restrict__ v_t,
    bf16* __restrict__ ctx)
{
    int bh = blockIdx.z, mt = blockIdx.y, cb = blockIdx.x;
    int tid = threadIdx.x, wave = tid >> 5, lane = tid & 31;
    int l15 = lane & 15, kb = (lane >> 4) * 8, hi8 = (lane >> 4) << 3;
    int b = bh >> 4, h = bh & 15;
    int ebase = cb * 512 + wave * 64;

    const bf16* arow = attn + ((size_t)bh * Nt + mt * 16 + l15) * Nt;
    v16bf af[3];
#pragma unroll
    for (int t = 0; t < 3; ++t)
        af[t] = cat8(*(const v8bf*)(arow + t * 32 + kb),
                     *(const v8bf*)(arow + t * 32 + 16 + kb));

    v16bf bfr[4][3];
#pragma unroll
    for (int s = 0; s < 4; ++s) {
        const bf16* brow = v_t + ((size_t)bh * (HD * Pp) + ebase + s * 16 + l15) * Nt;
#pragma unroll
        for (int t = 0; t < 3; ++t)
            bfr[s][t] = cat8(*(const v8bf*)(brow + t * 32 + kb),
                             *(const v8bf*)(brow + t * 32 + 16 + kb));
    }

    v8f acc[4];
#pragma unroll
    for (int s = 0; s < 4; ++s) acc[s] = (v8f){};
#pragma unroll
    for (int t = 0; t < 3; ++t)
#pragma unroll
        for (int s = 0; s < 4; ++s)
            acc[s] = wmma_bf16(af[t], bfr[s][t], acc[s]);

#pragma unroll
    for (int s = 0; s < 4; ++s) {
        int e = ebase + s * 16 + l15;      // e' = p*64 + d
        int p = e >> 6, d = e & 63;
#pragma unroll
        for (int j = 0; j < 8; ++j) {
            int ntok = mt * 16 + j + hi8;
            ctx[((size_t)((b * Nt + ntok) * Pp + p)) * Cc + h * HD + d] = (bf16)acc[s][j];
        }
    }
}

// ------------------------------------------------------------------ host ---
extern "C" void kernel_launch(void* const* d_in, const int* in_sizes, int n_in,
                              void* d_out, int out_size, void* d_ws, size_t ws_size,
                              hipStream_t stream)
{
    (void)in_sizes; (void)n_in; (void)out_size; (void)ws_size;
    const float* x     = (const float*)d_in[0];   // (4,96,64,1024)
    const float* Wqkv  = (const float*)d_in[1];   // (1024,3072)
    const float* Wproj = (const float*)d_in[2];   // (1024,1024)
    const float* bproj = (const float*)d_in[3];   // (1024,)
    float* out = (float*)d_out;

    uint8_t* ws = (uint8_t*)d_ws;
    size_t off = 0;
    auto alloc = [&](size_t bytes) -> void* {
        void* p = ws + off;
        off = (off + bytes + 255) & ~(size_t)255;
        return p;
    };

    bf16*  xb    = (bf16*)alloc((size_t)Mrows * Cc * 2);        // 50 MB (reused as ctx)
    bf16*  xp    = (bf16*)alloc((size_t)Bc * Nt * Cc * 2);      // pooled x
    bf16*  wqk_t = (bf16*)alloc((size_t)2048 * 1024 * 2);
    bf16*  wv_t  = (bf16*)alloc((size_t)1024 * 1024 * 2);
    bf16*  wp_t  = (bf16*)alloc((size_t)1024 * 1024 * 2);
    float* qk    = (float*)alloc((size_t)Bc * Nt * 2048 * 4);
    bf16*  v_t   = (bf16*)alloc((size_t)Bc * Hh * (HD * Pp) * Nt * 2);  // 50 MB
    bf16*  attn  = (bf16*)alloc((size_t)Bc * Hh * Nt * Nt * 2);
    bf16*  ctx   = xb;   // x_bf16 dead after v-projection; reuse

    // 1) pool over P + fp32->bf16 convert
    pool_convert_kernel<<<dim3(Cc / 256, Bc * Nt), 256, 0, stream>>>(x, xb, xp);

    // 2) weight transposes (B^T layout, bf16)
    transpose_conv_kernel<<<dim3(2048 / 32, 1024 / 32), dim3(32, 8), 0, stream>>>(
        Wqkv, 3072, 0, 1024, wqk_t);
    transpose_conv_kernel<<<dim3(1024 / 32, 1024 / 32), dim3(32, 8), 0, stream>>>(
        Wqkv, 3072, 2048, 1024, wv_t);
    transpose_conv_kernel<<<dim3(1024 / 32, 1024 / 32), dim3(32, 8), 0, stream>>>(
        Wproj, 1024, 0, 1024, wp_t);

    // 3) q/k pooled projection: (384 x 2048) = xp (384x1024) * wqk_t^T
    gemm_tiled_kernel<0><<<dim3(2048 / 128, 384 / 128), 256, 0, stream>>>(
        xp, wqk_t, Bc * Nt, 2048, 1024, qk, nullptr, nullptr);

    // 4) v projection: (24576 x 1024), epilogue scatters into v_t
    gemm_tiled_kernel<1><<<dim3(1024 / 128, Mrows / 128), 256, 0, stream>>>(
        xb, wv_t, Mrows, 1024, 1024, nullptr, v_t, nullptr);

    // 5) scores + softmax -> attn bf16
    scores_softmax_kernel<<<Bc * Hh, Nt, 0, stream>>>(qk, attn);

    // 6) attn @ v -> ctx (coalesced bf16, row-major (b,n,p) x C)
    attn_v_kernel<<<dim3(8, 6, Bc * Hh), 256, 0, stream>>>(attn, v_t, ctx);

    // 7) output projection + bias -> f32 d_out
    gemm_tiled_kernel<2><<<dim3(1024 / 128, Mrows / 128), 256, 0, stream>>>(
        ctx, wp_t, Mrows, 1024, 1024, out, nullptr, bproj);
}
